// MultiHeadAttention_2508260900975
// MI455X (gfx1250) — matrix operand order repaired and checked by reading, not compiled, not run on hardware
//
#include <hip/hip_runtime.h>

// Problem constants (fixed by the reference)
#define BQ   4
#define SQ   2048
#define HIDQ 1024
#define HQ   8
#define DQ   128

typedef __attribute__((ext_vector_type(16))) __bf16 bf16x16;
typedef __attribute__((ext_vector_type(8)))  float  floatx8;

__device__ __forceinline__ unsigned short f2bf(float f) {
    // round-to-nearest-even f32 -> bf16
    unsigned int u = __float_as_uint(f);
    unsigned int r = u + 0x7FFFu + ((u >> 16) & 1u);
    return (unsigned short)(r >> 16);
}

__device__ __forceinline__ unsigned pack2bf(float a, float b) {
    return (unsigned)f2bf(a) | ((unsigned)f2bf(b) << 16);
}

union Frag32B { bf16x16 v; uint4 q[2]; };

// A-fragment (16x32 bf16): lane l holds row M=l&15; kb = (l>=16)*16.
// VGPR0-7 = K[kb..kb+15]  (two contiguous 16B chunks)
__device__ __forceinline__ bf16x16 load_frag_a(const unsigned short* rowPtr, int kb) {
    Frag32B f;
    f.q[0] = *(const uint4*)(rowPtr + kb);
    f.q[1] = *(const uint4*)(rowPtr + kb + 8);
    return f.v;
}

// B-fragment (32x16 bf16): lane l holds column N=l&15; contiguous K range
// of 16 starting at kb2 = (l>=16)*16  (one 32B chunk)
__device__ __forceinline__ bf16x16 load_frag_b(const unsigned short* rowPtr, int kb2) {
    Frag32B f;
    f.q[0] = *(const uint4*)(rowPtr + kb2);
    f.q[1] = *(const uint4*)(rowPtr + kb2 + 8);
    return f.v;
}

#define WMMA_BF16(A, B, C) \
    __builtin_amdgcn_wmma_f32_16x16x32_bf16(false, (A), false, (B), (short)0, (C), false, false)

// CDNA5 async global->LDS copy, 16B per lane, tracked by ASYNCcnt.
// Generic LDS pointer truncates to the LDS offset (aperture keeps it in [31:0]).
__device__ __forceinline__ void async_load_b128(const unsigned short* gptr,
                                                unsigned short* lptr) {
    unsigned lds = (unsigned)(unsigned long long)lptr;
    asm volatile("global_load_async_to_lds_b128 %0, %1, off"
                 :: "v"(lds), "v"(gptr) : "memory");
}
__device__ __forceinline__ void wait_async0() {
    asm volatile("s_wait_asynccnt 0x0" ::: "memory");
}

// ---------------------------------------------------------------------------
// Transpose + convert weights: Wt[n*1024 + k] = bf16(W[k*1024 + n] * scale)
// ---------------------------------------------------------------------------
__global__ __launch_bounds__(256)
void prep_weight_kernel(const float* __restrict__ W, unsigned short* __restrict__ Wt,
                        float scale) {
    int idx = blockIdx.x * 256 + threadIdx.x;   // idx = n*1024 + k
    int n = idx >> 10, k = idx & 1023;
    Wt[idx] = f2bf(W[k * 1024 + n] * scale);
}

// ---------------------------------------------------------------------------
// Tiled WMMA GEMM (used for the 4 dense projections):
//   C[128x64 tile] = A @ Bt^T, A row-major [M,K] (f32 converted or bf16-async),
//   Bt n-major [N,K] bf16 (async staged), C f32 or bf16.
// 256 threads = 8 waves; wave (wm,wn) owns a 32x32 block = 2x2 WMMA tiles.
// ---------------------------------------------------------------------------
template<bool A_F32, bool OUT_F32>
__global__ __launch_bounds__(256)
void wmma_gemm_kernel(const void* __restrict__ Aptr,
                      const unsigned short* __restrict__ Bptr,
                      void* __restrict__ Cptr,
                      int K, long lda, long ldb, long ldc) {
    __shared__ __align__(16) unsigned short ldsA[128 * 32];
    __shared__ __align__(16) unsigned short ldsB[64 * 32];

    const float*          Af = (const float*)Aptr;
    const unsigned short* Au = (const unsigned short*)Aptr;

    const int tid  = threadIdx.x;
    const int lane = tid & 31;
    const int w    = tid >> 5;
    const int wm   = w & 3;           // 0..3 : 32-row strip in tile
    const int wn   = w >> 2;          // 0..1 : 32-col strip in tile
    const int lr   = lane & 15;
    const int kbA  = (lane >> 4) * 16;
    const int kbB  = (lane >> 4) * 16;

    const long row0 = (long)blockIdx.y * 128;
    const long col0 = (long)blockIdx.x * 64;

    floatx8 zero = {};
    floatx8 acc00 = zero, acc01 = zero, acc10 = zero, acc11 = zero;

    for (int kk = 0; kk < K; kk += 32) {
        if (A_F32) {
            // 128x32 f32 tile: float4 loads, packed-bf16 b64 stores
            #pragma unroll
            for (int i = 0; i < 4; ++i) {
                int e4 = (tid + i * 256) * 4;
                int r = e4 >> 5, c = e4 & 31;
                float4 fv = *(const float4*)&Af[(row0 + r) * lda + kk + c];
                uint2 pk;
                pk.x = pack2bf(fv.x, fv.y);
                pk.y = pack2bf(fv.z, fv.w);
                *(uint2*)&ldsA[e4] = pk;
            }
        } else {
            // 128x32 bf16 tile: async global->LDS, 16B per lane
            #pragma unroll
            for (int i = 0; i < 2; ++i) {
                int e8 = (tid + i * 256) * 8;
                int r = e8 >> 5, c = e8 & 31;
                async_load_b128(&Au[(row0 + r) * lda + kk + c], &ldsA[e8]);
            }
        }
        {
            // 64x32 bf16 B tile: async global->LDS
            int e8 = tid * 8;
            int n = e8 >> 5, c = e8 & 31;
            async_load_b128(&Bptr[(col0 + n) * ldb + kk + c], &ldsB[e8]);
        }
        wait_async0();
        __syncthreads();

        bf16x16 a0 = load_frag_a(&ldsA[(wm * 32 +  0 + lr) * 32], kbA);
        bf16x16 a1 = load_frag_a(&ldsA[(wm * 32 + 16 + lr) * 32], kbA);
        bf16x16 b0 = load_frag_b(&ldsB[(wn * 32 +  0 + lr) * 32], kbB);
        bf16x16 b1 = load_frag_b(&ldsB[(wn * 32 + 16 + lr) * 32], kbB);

        acc00 = WMMA_BF16(a0, b0, acc00);
        acc01 = WMMA_BF16(a0, b1, acc01);
        acc10 = WMMA_BF16(a1, b0, acc10);
        acc11 = WMMA_BF16(a1, b1, acc11);
        __syncthreads();
    }

    float*          Cf = (float*)Cptr;
    unsigned short* Cu = (unsigned short*)Cptr;
    const int mhalf = (lane >> 4) * 8;

    floatx8 accs[2][2] = {{acc00, acc01}, {acc10, acc11}};
    #pragma unroll
    for (int tm = 0; tm < 2; ++tm) {
        #pragma unroll
        for (int tn = 0; tn < 2; ++tn) {
            long n = col0 + wn * 32 + tn * 16 + lr;
            #pragma unroll
            for (int r = 0; r < 8; ++r) {
                long m = row0 + wm * 32 + tm * 16 + mhalf + r;
                if (OUT_F32) Cf[m * ldc + n] = accs[tm][tn][r];
                else         Cu[m * ldc + n] = f2bf(accs[tm][tn][r]);
            }
        }
    }
}

// ---------------------------------------------------------------------------
// Fused attention: per (b, h, 128-query block):
//   Pass A: online row max/sum over masked Q@K^T (no score storage)
//   Pass B: recompute scores, emit normalized probs to attn (single HBM write)
//           and to wave-private LDS (bf16), accumulate ctx += P @ V with WMMA.
// 8 waves x 16 query rows. Q fragments kept in registers for both passes.
// K tiles staged via async global->LDS; V tiles transposed on store.
// ---------------------------------------------------------------------------
__global__ __launch_bounds__(256)
void fused_attn_kernel(const unsigned short* __restrict__ Qp,
                       const unsigned short* __restrict__ Kp,
                       const unsigned short* __restrict__ Vp,
                       const int* __restrict__ mask,
                       float* __restrict__ attn,
                       unsigned short* __restrict__ Ctx) {
    __shared__ __align__(16) unsigned short ldsK[64 * 128];      // [key][d]
    __shared__ __align__(16) unsigned short ldsV[128 * 64];      // [d][key] (transposed)
    __shared__ __align__(16) unsigned short ldsP[8 * 16 * 64];   // per-wave P tiles

    const int z  = blockIdx.y;          // b*H + h
    const int bb = z >> 3;
    const int hh = z & 7;
    const long qb0 = (long)blockIdx.x * 128;

    const long headOff = (long)bb * SQ * HIDQ + (long)hh * DQ;
    const unsigned short* QpH = Qp + headOff;
    const unsigned short* KpH = Kp + headOff;
    const unsigned short* VpH = Vp + headOff;
    const int* mrow = mask + bb * SQ;
    float* attnZ = attn + (long)z * SQ * SQ;

    const int tid   = threadIdx.x;
    const int lane  = tid & 31;
    const int w     = tid >> 5;
    const int lr    = lane & 15;
    const int kbA   = (lane >> 4) * 16;
    const int kbB   = (lane >> 4) * 16;
    const int mhalf = (lane >> 4) * 8;

    // This wave's 16 Q rows as 4 A-fragments (one per 32-wide k-step), in regs
    bf16x16 qf[4];
    {
        const unsigned short* qrow = QpH + (qb0 + w * 16 + lr) * HIDQ;
        #pragma unroll
        for (int ks = 0; ks < 4; ++ks) qf[ks] = load_frag_a(qrow + ks * 32, kbA);
    }

    float rowmax[8], rowsum[8];
    #pragma unroll
    for (int r = 0; r < 8; ++r) { rowmax[r] = -3.0e38f; rowsum[r] = 0.0f; }

    floatx8 zero = {};

    // ----------------- Pass A: online softmax statistics --------------------
    for (int kb = 0; kb < 32; ++kb) {
        const long kb0 = (long)kb * 64;
        __syncthreads();
        #pragma unroll
        for (int i = 0; i < 4; ++i) {
            int e8 = (tid + i * 256) * 8;
            int k2 = e8 >> 7, d = e8 & 127;
            async_load_b128(&KpH[(kb0 + k2) * HIDQ + d], &ldsK[e8]);
        }
        if (kb < 31) __builtin_prefetch(&KpH[(kb0 + 64 + (tid >> 3)) * HIDQ + (tid & 7) * 16], 0, 1);
        wait_async0();
        __syncthreads();

        floatx8 sacc[4] = {zero, zero, zero, zero};
        #pragma unroll
        for (int tn = 0; tn < 4; ++tn) {
            #pragma unroll
            for (int ks = 0; ks < 4; ++ks) {
                bf16x16 bf = load_frag_b(&ldsK[(tn * 16 + lr) * 128 + ks * 32], kbB);
                sacc[tn] = WMMA_BF16(qf[ks], bf, sacc[tn]);
            }
            int mk = mrow[kb0 + tn * 16 + lr];
            if (mk) {
                #pragma unroll
                for (int r = 0; r < 8; ++r) sacc[tn][r] = -1e9f;
            }
        }
        #pragma unroll
        for (int r = 0; r < 8; ++r) {
            float bm = fmaxf(fmaxf(sacc[0][r], sacc[1][r]), fmaxf(sacc[2][r], sacc[3][r]));
            #pragma unroll
            for (int o = 8; o > 0; o >>= 1) bm = fmaxf(bm, __shfl_xor(bm, o, 32));
            float mnew = fmaxf(rowmax[r], bm);
            float corr = __expf(rowmax[r] - mnew);
            float ps = __expf(sacc[0][r] - mnew) + __expf(sacc[1][r] - mnew)
                     + __expf(sacc[2][r] - mnew) + __expf(sacc[3][r] - mnew);
            #pragma unroll
            for (int o = 8; o > 0; o >>= 1) ps += __shfl_xor(ps, o, 32);
            rowsum[r] = rowsum[r] * corr + ps;
            rowmax[r] = mnew;
        }
    }

    float invsum[8];
    #pragma unroll
    for (int r = 0; r < 8; ++r) invsum[r] = 1.0f / rowsum[r];

    // ----- Pass B: normalized probs -> attn (HBM, once) + ctx = P @ V -------
    floatx8 cacc[8];
    #pragma unroll
    for (int t = 0; t < 8; ++t) cacc[t] = zero;

    for (int kb = 0; kb < 32; ++kb) {
        const long kb0 = (long)kb * 64;
        __syncthreads();
        #pragma unroll
        for (int i = 0; i < 4; ++i) {
            int e8 = (tid + i * 256) * 8;
            int k2 = e8 >> 7, d = e8 & 127;
            async_load_b128(&KpH[(kb0 + k2) * HIDQ + d], &ldsK[e8]);
            // V tile: b128 load + transpose scatter into [d][key]
            uint4 vv = *(const uint4*)&VpH[(kb0 + k2) * HIDQ + d];
            const unsigned short* vs = (const unsigned short*)&vv;
            #pragma unroll
            for (int j = 0; j < 8; ++j) ldsV[(d + j) * 64 + k2] = vs[j];
        }
        if (kb < 31) __builtin_prefetch(&VpH[(kb0 + 64 + (tid >> 3)) * HIDQ + (tid & 7) * 16], 0, 1);
        wait_async0();
        __syncthreads();

        floatx8 sacc[4] = {zero, zero, zero, zero};
        #pragma unroll
        for (int tn = 0; tn < 4; ++tn) {
            #pragma unroll
            for (int ks = 0; ks < 4; ++ks) {
                bf16x16 bf = load_frag_b(&ldsK[(tn * 16 + lr) * 128 + ks * 32], kbB);
                sacc[tn] = WMMA_BF16(qf[ks], bf, sacc[tn]);
            }
        }

        unsigned short* Pw = &ldsP[w * 1024];
        #pragma unroll
        for (int tn = 0; tn < 4; ++tn) {
            int mk = mrow[kb0 + tn * 16 + lr];
            #pragma unroll
            for (int r = 0; r < 8; ++r) {
                float s = mk ? -1e9f : sacc[tn][r];
                float p = __expf(s - rowmax[r]) * invsum[r];
                attnZ[(qb0 + w * 16 + mhalf + r) * SQ + kb0 + tn * 16 + lr] = p;
                Pw[(mhalf + r) * 64 + tn * 16 + lr] = f2bf(p);  // wave-private, in-order DS
            }
        }

        // ctx(16x128) += P(16x64) @ V(64x128); V staged transposed -> contiguous B-frags
        #pragma unroll
        for (int ks = 0; ks < 2; ++ks) {
            bf16x16 af = load_frag_a(&Pw[lr * 64 + ks * 32], kbA);
            #pragma unroll
            for (int tn = 0; tn < 8; ++tn) {
                bf16x16 bf = load_frag_b(&ldsV[(tn * 16 + lr) * 64 + ks * 32], kbB);
                cacc[tn] = WMMA_BF16(af, bf, cacc[tn]);
            }
        }
    }

    // Epilogue: ctx -> bf16, merged-head layout [B,S,HID]
    unsigned short* CtxH = Ctx + headOff;
    #pragma unroll
    for (int tn = 0; tn < 8; ++tn) {
        #pragma unroll
        for (int r = 0; r < 8; ++r) {
            CtxH[(qb0 + w * 16 + mhalf + r) * HIDQ + tn * 16 + lr] = f2bf(cacc[tn][r]);
        }
    }
}

// ---------------------------------------------------------------------------
extern "C" void kernel_launch(void* const* d_in, const int* in_sizes, int n_in,
                              void* d_out, int out_size, void* d_ws, size_t ws_size,
                              hipStream_t stream) {
    (void)in_sizes; (void)n_in; (void)out_size; (void)ws_size;

    const float* q    = (const float*)d_in[0];
    const float* k    = (const float*)d_in[1];
    const float* v    = (const float*)d_in[2];
    const int*   mask = (const int*)d_in[3];
    const float* Wq   = (const float*)d_in[4];
    const float* Wk   = (const float*)d_in[5];
    const float* Wv   = (const float*)d_in[6];
    const float* Wo   = (const float*)d_in[7];

    float* out  = (float*)d_out;                       // [B,S,HID]
    float* attn = out + (long)BQ * SQ * HIDQ;          // [B,H,S,S]

    // workspace (bf16 as ushort): 4 transposed weights, Qp, Kp, Vp, Ctx (~72MB)
    unsigned short* Wtq = (unsigned short*)d_ws;
    unsigned short* Wtk = Wtq + 1024 * 1024;
    unsigned short* Wtv = Wtk + 1024 * 1024;
    unsigned short* Wto = Wtv + 1024 * 1024;
    unsigned short* Qp  = Wto + 1024 * 1024;
    unsigned short* Kp  = Qp + (long)BQ * SQ * HIDQ;
    unsigned short* Vp  = Kp + (long)BQ * SQ * HIDQ;
    unsigned short* Ctx = Vp + (long)BQ * SQ * HIDQ;

    const float scale = 0.08838834764831845f;          // D^-0.5, folded into Wq

    prep_weight_kernel<<<4096, 256, 0, stream>>>(Wq, Wtq, scale);
    prep_weight_kernel<<<4096, 256, 0, stream>>>(Wk, Wtk, 1.0f);
    prep_weight_kernel<<<4096, 256, 0, stream>>>(Wv, Wtv, 1.0f);
    prep_weight_kernel<<<4096, 256, 0, stream>>>(Wo, Wto, 1.0f);

    // Projections: [8192,1024] @ [1024,1024] -> bf16
    dim3 gProj(HIDQ / 64, (BQ * SQ) / 128, 1);
    wmma_gemm_kernel<true, false><<<gProj, 256, 0, stream>>>(
        q, Wtq, Qp, HIDQ, HIDQ, HIDQ, HIDQ);
    wmma_gemm_kernel<true, false><<<gProj, 256, 0, stream>>>(
        k, Wtk, Kp, HIDQ, HIDQ, HIDQ, HIDQ);
    wmma_gemm_kernel<true, false><<<gProj, 256, 0, stream>>>(
        v, Wtv, Vp, HIDQ, HIDQ, HIDQ, HIDQ);

    // Fused scores + softmax + attn-write + ctx
    dim3 gAttn(SQ / 128, BQ * HQ, 1);
    fused_attn_kernel<<<gAttn, 256, 0, stream>>>(Qp, Kp, Vp, mask, attn, Ctx);

    // Output projection: [8192,1024] @ Wo -> f32 out
    wmma_gemm_kernel<false, true><<<gProj, 256, 0, stream>>>(
        Ctx, Wto, out, HIDQ, HIDQ, HIDQ, HIDQ);
}
